// TorchROIPooling_47665547051598
// MI455X (gfx1250) — matrix operand order repaired and checked by reading, not compiled, not run on hardware
//
#include <hip/hip_runtime.h>
#include <cstdint>
#include <cstddef>

// ============================================================================
// MI455X (gfx1250) implementation.
// - All GEMM-shaped math (convs via implicit GEMM, FC layers) runs on
//   v_wmma_f32_16x16x32_bf16 (wave32, one 16x16 tile per wave).
// - Inter-layer activations are carried as bf16 (BN renormalizes each layer),
//   halving HBM traffic on the memory-bound big tensors.
// - Conv im2col A-fragments are gathered from an LDS-staged halo tile via a
//   precomputed k->LDS-offset lookup table (no per-element divisions); an
//   appended zero pad makes OOB taps unconditional reads.
// - A fragment is built once per K-step and reused across all NT out-channel
//   tiles (nt loop inside the kernel).
// - Training-mode BN stats via wave-shuffle + LDS + global f32 atomics.
// Workspace assumption: ws_size >= ~300 MB (268MB f32 conv scratch + bf16
// activations + bf16 weights).
// ============================================================================

typedef __attribute__((ext_vector_type(16))) __bf16 bf16x16;
typedef __attribute__((ext_vector_type(8)))  float  v8f;

union FragBF {
  bf16x16        v;
  unsigned short s[16];
  uint4          q[2];
};

__device__ __forceinline__ unsigned short f2bf(float f) {
  unsigned u = __float_as_uint(f);
  u += 0x7FFFu + ((u >> 16) & 1u);          // round-to-nearest-even
  return (unsigned short)(u >> 16);
}
__device__ __forceinline__ unsigned short to_bf(float f)          { return f2bf(f); }
__device__ __forceinline__ unsigned short to_bf(unsigned short h) { return h; }

// ---------------------------------------------------------------------------
// Utility kernels
// ---------------------------------------------------------------------------
__global__ void k_zero(float* p, int n) {
  int i = blockIdx.x * blockDim.x + threadIdx.x;
  if (i < n) p[i] = 0.0f;
}

__global__ void k_f2bf(const float* __restrict__ src, unsigned short* __restrict__ dst, int n) {
  int i = blockIdx.x * blockDim.x + threadIdx.x;
  if (i < n) dst[i] = f2bf(src[i]);
}

// Repack conv weights [CO][CIN][27] f32 -> [CO][KSTEPS*32] bf16 with
// k = c*28 + off (off 27 and any c >= CIN zero-padded).
template<int CO, int CIN, int KSTEPS>
__global__ void k_pack_w(const float* __restrict__ w, unsigned short* __restrict__ wbf) {
  const int K = KSTEPS * 32;
  int i = blockIdx.x * blockDim.x + threadIdx.x;
  if (i >= CO * K) return;
  int n = i / K, k = i % K;
  int c = k / 28, off = k % 28;
  unsigned short v = 0;
  if (c < CIN && off < 27) v = f2bf(w[(n * CIN + c) * 27 + off]);
  wbf[i] = v;
}

// ---------------------------------------------------------------------------
// Implicit-GEMM 3x3x3 SAME conv via WMMA bf16.
//   A tile: M = 16 consecutive x positions at fixed (b,z,y); K = CIN*28 (pad).
//   B tile: N = 16 out channels (looped NT times); weights pre-packed
//   [CO][KSTEPS*32] bf16 (k-contiguous -> 32B contiguous B-fragment loads).
// One wave (32 threads) per block. LDS stages the [CIN][3][3][18] halo plus a
// 16-entry zero pad, and a k -> tile-offset lookup table (sentinel -> pad).
// ---------------------------------------------------------------------------
template<typename TIN, int CIN, int DIM, int CO, int KSTEPS>
__global__ void k_conv_wmma(const TIN* __restrict__ src,              // [16][CIN][DIM]^3
                            const unsigned short* __restrict__ wbf,   // [CO][KSTEPS*32]
                            const float* __restrict__ bias,           // [CO]
                            float* __restrict__ dst)                  // [16][CO][DIM]^3
{
  constexpr int XT   = DIM / 16;
  constexpr int NT   = CO / 16;
  constexpr int TILE = CIN * 3 * 3 * 18;
  constexpr int KTOT = KSTEPS * 32;

  int bid = blockIdx.x;
  int xt = bid % XT; bid /= XT;
  int y  = bid % DIM; bid /= DIM;
  int z  = bid % DIM; bid /= DIM;
  int b  = bid;
  int lane = threadIdx.x;

  __shared__ unsigned short tile[TILE + 16];   // +16 zeros: OOB-tap landing pad
  __shared__ unsigned short ldsoff[KTOT];      // k -> offset into tile (or TILE)

  const int x0 = xt * 16;
  for (int idx = lane; idx < TILE + 16; idx += 32) {
    unsigned short v = 0;
    if (idx < TILE) {
      int dx = idx % 18;
      int t  = idx / 18;
      int dy = t % 3; t /= 3;
      int dz = t % 3;
      int c  = t / 3;
      int zz = z + dz - 1, yy = y + dy - 1, xx = x0 + dx - 1;
      if ((unsigned)zz < (unsigned)DIM && (unsigned)yy < (unsigned)DIM &&
          (unsigned)xx < (unsigned)DIM) {
        size_t o = (((size_t)(b * CIN + c) * DIM + zz) * DIM + yy) * DIM + xx;
        v = to_bf(src[o]);
      }
    }
    tile[idx] = v;
  }
  // k -> LDS offset table (shared across the whole K loop; kills the
  // per-element /28, /9, /3 VALU work in the hot gather).
  for (int k = lane; k < KTOT; k += 32) {
    int c = k / 28, off = k % 28;
    unsigned short t = (unsigned short)TILE;   // zero pad
    if (c < CIN && off < 27) {
      int kd = off / 9, r = off - kd * 9;
      int kh = r / 3,  kw = r - kh * 3;
      t = (unsigned short)(((c * 3 + kd) * 3 + kh) * 18 + kw);
    }
    ldsoff[k] = t;
  }
  __syncthreads();

  const int m     = lane & 15;     // A row (spatial x) == B column source
  const int khalf = lane >> 4;

  v8f acc[NT] = {};

  for (int kk = 0; kk < KSTEPS; ++kk) {
    // A fragment: contiguous K per lane-half (matches B).
    // Element j holds K = kk*32 + khalf*16 + j.
    FragBF a;
    const unsigned short* tb = &ldsoff[kk * 32 + khalf * 16];
    const unsigned short* tm = &tile[m];
#pragma unroll
    for (int j = 0; j < 16; ++j)
      a.s[j] = tm[tb[j]];

#pragma unroll
    for (int nt = 0; nt < NT; ++nt) {
      // B fragment: contiguous K per lane-half -> one 32B run of weight row n.
      const unsigned short* wrow =
          wbf + (size_t)(nt * 16 + m) * KTOT + kk * 32 + khalf * 16;
      __builtin_prefetch((const void*)(wrow + 32), 0, 1);
      FragBF bf;
      bf.q[0] = *(const uint4*)(wrow);
      bf.q[1] = *(const uint4*)(wrow + 8);
      acc[nt] = __builtin_amdgcn_wmma_f32_16x16x32_bf16(
          false, a.v, false, bf.v, (short)0, acc[nt], false, false);
    }
  }

#pragma unroll
  for (int nt = 0; nt < NT; ++nt) {
    const int n  = nt * 16 + m;
    const float bv = bias[n];
    const int mrow = khalf * 8;
#pragma unroll
    for (int r = 0; r < 8; ++r) {
      int xo = x0 + mrow + r;
      size_t o = (((size_t)(b * CO + n) * DIM + z) * DIM + y) * DIM + xo;
      dst[o] = acc[nt][r] + bv;
    }
  }
}

// ---------------------------------------------------------------------------
// Per-channel sum / sumsq over [B][C][S] with wave shuffles + atomics.
// grid = (C, CHUNKS), block = 256.
// ---------------------------------------------------------------------------
__global__ void k_stats(const float* __restrict__ y, float* __restrict__ sum,
                        float* __restrict__ sq, int B, int C, int S) {
  int c = blockIdx.x;
  int chunk = blockIdx.y, nch = gridDim.y;
  float s = 0.f, s2 = 0.f;
  for (int b = 0; b < B; ++b) {
    const float* p = y + ((size_t)b * C + c) * S;
    for (int i = chunk * blockDim.x + threadIdx.x; i < S; i += blockDim.x * nch) {
      float v = p[i];
      s += v; s2 += v * v;
    }
  }
  for (int o = 16; o > 0; o >>= 1) {
    s  += __shfl_down(s, o, 32);
    s2 += __shfl_down(s2, o, 32);
  }
  __shared__ float ls[16];
  int w = threadIdx.x >> 5;
  if ((threadIdx.x & 31) == 0) { ls[w] = s; ls[8 + w] = s2; }
  __syncthreads();
  if (threadIdx.x == 0) {
    float ts = 0.f, ts2 = 0.f;
    for (int i = 0; i < (int)(blockDim.x >> 5); ++i) { ts += ls[i]; ts2 += ls[8 + i]; }
    atomicAdd(&sum[c], ts);
    atomicAdd(&sq[c], ts2);
  }
}

__global__ void k_bnparams(const float* __restrict__ sum, const float* __restrict__ sq,
                           const float* __restrict__ g, const float* __restrict__ beta,
                           float* __restrict__ scale, float* __restrict__ shift,
                           int C, float invN) {
  int c = blockIdx.x * blockDim.x + threadIdx.x;
  if (c >= C) return;
  float m  = sum[c] * invN;
  float vv = sq[c] * invN - m * m;
  float sc = g[c] * rsqrtf(vv + 1e-5f);
  scale[c] = sc;
  shift[c] = beta[c] - m * sc;
}

// Fused BN + ReLU + 2x2x2 maxpool, writes bf16. out: [16][C][DIN/2]^3
template<int C, int DIN>
__global__ void k_bnrelupool(const float* __restrict__ y, const float* __restrict__ scale,
                             const float* __restrict__ shift, unsigned short* __restrict__ out) {
  constexpr int DO = DIN / 2;
  int i = blockIdx.x * blockDim.x + threadIdx.x;
  if (i >= 16 * C * DO * DO * DO) return;
  int x = i % DO; int t = i / DO;
  int yy = t % DO; t /= DO;
  int z  = t % DO; t /= DO;
  int c  = t % C;
  int b  = t / C;
  float sc = scale[c], sh = shift[c];
  const float* base = y + (((size_t)(b * C + c) * DIN + z * 2) * DIN + yy * 2) * DIN + x * 2;
  float mx = -__builtin_inff();
#pragma unroll
  for (int dz = 0; dz < 2; ++dz)
#pragma unroll
    for (int dy = 0; dy < 2; ++dy)
#pragma unroll
      for (int dx = 0; dx < 2; ++dx) {
        float v = fmaxf(base[(dz * DIN + dy) * DIN + dx] * sc + sh, 0.f);
        mx = fmaxf(mx, v);
      }
  out[i] = f2bf(mx);
}

// BN only, elementwise in-place f32 (conv3 output feeds ROI pool in f32)
__global__ void k_bn_inplace(float* __restrict__ y, const float* __restrict__ scale,
                             const float* __restrict__ shift, int C, int S, int total) {
  int i = blockIdx.x * blockDim.x + threadIdx.x;
  if (i >= total) return;
  int c = (i / S) % C;
  y[i] = y[i] * scale[c] + shift[c];
}

// ---------------------------------------------------------------------------
// ROI adaptive-avg pool (OUT=4) on feat [16][64][16][16][16] -> bf16 [16][4096]
// ---------------------------------------------------------------------------
__global__ void k_roipool(const float* __restrict__ h, const float* __restrict__ prop,
                          unsigned short* __restrict__ A0) {
  int i = blockIdx.x * blockDim.x + threadIdx.x;
  if (i >= 16 * 64 * 64) return;
  int k  = i & 3;
  int j  = (i >> 2) & 3;
  int ii = (i >> 4) & 3;
  int c  = (i >> 6) & 63;
  int b  = i >> 12;

  const float* p = prop + b * 6;
  float f0 = floorf(p[0] * 0.242f), f1 = floorf(p[1] * 0.242f), f2 = floorf(p[2] * 0.242f);
  float c3 = ceilf(p[3] * 0.242f),  c4 = ceilf(p[4] * 0.242f),  c5 = ceilf(p[5] * 0.242f);
  bool zero = (f0 == 0.f) && (f1 == 0.f) && (f2 == 0.f) &&
              (c3 == 0.f) && (c4 == 0.f) && (c5 == 0.f);
  int x0 = (int)f0, y0 = (int)f1, z0 = (int)f2;
  int x1 = min((int)c3 + 1, 16), y1 = min((int)c4 + 1, 16), z1 = min((int)c5 + 1, 16);
  int Lx = x1 - x0, Ly = y1 - y0, Lz = z1 - z0;
  int sx = x0 + (ii * Lx) / 4, ex = x0 + ((ii + 1) * Lx + 3) / 4;
  int sy = y0 + (j  * Ly) / 4, ey = y0 + ((j  + 1) * Ly + 3) / 4;
  int sz = z0 + (k  * Lz) / 4, ez = z0 + ((k  + 1) * Lz + 3) / 4;

  const float* hb = h + (size_t)(b * 64 + c) * 4096;
  float s = 0.f;
  for (int xx = sx; xx < ex; ++xx)
    for (int yy = sy; yy < ey; ++yy)
      for (int zz = sz; zz < ez; ++zz)
        s += hb[(xx * 16 + yy) * 16 + zz];
  int cnt = (ex - sx) * (ey - sy) * (ez - sz);
  float r = (zero || cnt <= 0) ? 0.f : s / (float)cnt;
  A0[i] = f2bf(r);
}

// ---------------------------------------------------------------------------
// FC via WMMA: out[16][N] = A[16][K](bf16) x W[N][K](bf16)^T + bias. One wave
// per 16-column tile; M=16 = exactly one WMMA tile of batch rows.
// ---------------------------------------------------------------------------
__global__ void k_fc_wmma(const unsigned short* __restrict__ A,
                          const unsigned short* __restrict__ W,
                          const float* __restrict__ bias,
                          float* __restrict__ out, int K, int N) {
  int lane  = threadIdx.x;
  int col   = lane & 15;
  int khalf = lane >> 4;
  int n     = blockIdx.x * 16 + col;

  const unsigned short* arow = A + (size_t)col * K;   // A row m == lane&15
  const unsigned short* wrow = W + (size_t)n * K;

  v8f acc = {};
  for (int kk = 0; kk < K; kk += 32) {
    __builtin_prefetch((const void*)(wrow + kk + 128), 0, 1);
    FragBF a, bf;
    a.q[0]  = *(const uint4*)(arow + kk + khalf * 16);       // K = khalf*16 + 0..7
    a.q[1]  = *(const uint4*)(arow + kk + khalf * 16 + 8);   // K = khalf*16 + 8..15
    bf.q[0] = *(const uint4*)(wrow + kk + khalf * 16);       // K = khalf*16 + 0..15
    bf.q[1] = *(const uint4*)(wrow + kk + khalf * 16 + 8);
    acc = __builtin_amdgcn_wmma_f32_16x16x32_bf16(
        false, a.v, false, bf.v, (short)0, acc, false, false);
  }
  float bv = bias[n];
  int mbase = khalf * 8;
#pragma unroll
  for (int r = 0; r < 8; ++r)
    out[(size_t)(mbase + r) * N + n] = acc[r] + bv;
}

// Per-feature BN (batch of 16) + ReLU, writes next bf16 activation [16][N]
__global__ void k_fcbn(const float* __restrict__ y, const float* __restrict__ g,
                       const float* __restrict__ beta, unsigned short* __restrict__ A, int N) {
  int n = blockIdx.x * blockDim.x + threadIdx.x;
  if (n >= N) return;
  float s = 0.f, s2 = 0.f;
#pragma unroll 4
  for (int b = 0; b < 16; ++b) {
    float v = y[b * N + n];
    s += v; s2 += v * v;
  }
  float m  = s * 0.0625f;
  float vv = s2 * 0.0625f - m * m;
  float sc = g[n] * rsqrtf(vv + 1e-5f);
  float sh = beta[n] - m * sc;
#pragma unroll 4
  for (int b = 0; b < 16; ++b) {
    float v = y[b * N + n] * sc + sh;
    A[b * N + n] = f2bf(fmaxf(v, 0.f));
  }
}

// ---------------------------------------------------------------------------
// Host launcher
// ---------------------------------------------------------------------------
extern "C" void kernel_launch(void* const* d_in, const int* in_sizes, int n_in,
                              void* d_out, int out_size, void* d_ws, size_t ws_size,
                              hipStream_t stream) {
  (void)in_sizes; (void)n_in; (void)out_size; (void)ws_size;
  // setup_inputs() dict order: x, proposals, then params in insertion order.
  const float* x     = (const float*)d_in[0];   // [16,1,64,64,64]
  const float* prop  = (const float*)d_in[1];   // [16,1,6]
  const float* c1w   = (const float*)d_in[2];
  const float* c1b   = (const float*)d_in[3];
  const float* bn1g  = (const float*)d_in[4];
  const float* bn1b  = (const float*)d_in[5];
  const float* c2w   = (const float*)d_in[6];
  const float* c2b   = (const float*)d_in[7];
  const float* bn2g  = (const float*)d_in[8];
  const float* bn2b  = (const float*)d_in[9];
  const float* c3w   = (const float*)d_in[10];
  const float* c3b   = (const float*)d_in[11];
  const float* bn3g  = (const float*)d_in[12];
  const float* bn3b  = (const float*)d_in[13];
  const float* fcw[5], *fcb[5], *fbg[4], *fbb[4];
  int idx = 14;
  for (int i = 0; i < 5; ++i) {
    fcw[i] = (const float*)d_in[idx++];
    fcb[i] = (const float*)d_in[idx++];
    if (i < 4) {
      fbg[i] = (const float*)d_in[idx++];
      fbb[i] = (const float*)d_in[idx++];
    }
  }

  char* ws = (char*)d_ws;
  size_t off = 0;
  auto take = [&](size_t bytes) -> void* {
    void* p = ws + off;
    off += (bytes + 255) & ~(size_t)255;
    return p;
  };
  float*          BIG   = (float*)take((size_t)16 * 16 * 262144 * 4);  // y1/y2/y3 (f32)
  unsigned short* ACT   = (unsigned short*)take((size_t)16 * 16 * 32768 * 2); // a1/a2 bf16
  unsigned short* w1bf  = (unsigned short*)take(16 * 32 * 2);
  unsigned short* w2bf  = (unsigned short*)take(32 * 448 * 2);
  unsigned short* w3bf  = (unsigned short*)take(64 * 896 * 2);
  unsigned short* fw0   = (unsigned short*)take((size_t)1024 * 4096 * 2);
  unsigned short* fw1   = (unsigned short*)take((size_t)512 * 1024 * 2);
  unsigned short* fw2   = (unsigned short*)take(256 * 512 * 2);
  unsigned short* fw3   = (unsigned short*)take(128 * 256 * 2);
  unsigned short* fw4   = (unsigned short*)take(64 * 128 * 2);
  float*          stats = (float*)take(128 * 4);   // sum[64] | sq[64]
  float*          bnp   = (float*)take(128 * 4);   // scale[64] | shift[64]
  unsigned short* A0    = (unsigned short*)take(16 * 4096 * 2);
  unsigned short* A1    = (unsigned short*)take(16 * 1024 * 2);
  unsigned short* A2    = (unsigned short*)take(16 * 512 * 2);
  unsigned short* A3    = (unsigned short*)take(16 * 256 * 2);
  unsigned short* A4    = (unsigned short*)take(16 * 128 * 2);
  float*          F     = (float*)take(16 * 1024 * 4);  // fc scratch (max N=1024)

  float* sum = stats, *sq = stats + 64;
  float* scale = bnp, *shift = bnp + 64;

  // ---- weight prep (every call; deterministic) ----
  k_pack_w<16, 1, 1><<<(16 * 32 + 255) / 256, 256, 0, stream>>>(c1w, w1bf);
  k_pack_w<32, 16, 14><<<(32 * 448 + 255) / 256, 256, 0, stream>>>(c2w, w2bf);
  k_pack_w<64, 32, 28><<<(64 * 896 + 255) / 256, 256, 0, stream>>>(c3w, w3bf);
  k_f2bf<<<(4194304 + 255) / 256, 256, 0, stream>>>(fcw[0], fw0, 4194304);
  k_f2bf<<<(524288 + 255) / 256, 256, 0, stream>>>(fcw[1], fw1, 524288);
  k_f2bf<<<(131072 + 255) / 256, 256, 0, stream>>>(fcw[2], fw2, 131072);
  k_f2bf<<<(32768 + 255) / 256, 256, 0, stream>>>(fcw[3], fw3, 32768);
  k_f2bf<<<(8192 + 255) / 256, 256, 0, stream>>>(fcw[4], fw4, 8192);

  // ---- stage 1: conv1 -> BN stats -> BN+ReLU+pool -> a1 bf16 [16,16,32^3]
  k_conv_wmma<float, 1, 64, 16, 1>
      <<<16 * 64 * 64 * 4, 32, 0, stream>>>(x, w1bf, c1b, BIG);
  k_zero<<<1, 128, 0, stream>>>(stats, 128);
  k_stats<<<dim3(16, 32), 256, 0, stream>>>(BIG, sum, sq, 16, 16, 262144);
  k_bnparams<<<1, 64, 0, stream>>>(sum, sq, bn1g, bn1b, scale, shift, 16,
                                   1.0f / (16.0f * 262144.0f));
  k_bnrelupool<16, 64>
      <<<(16 * 16 * 32768 + 255) / 256, 256, 0, stream>>>(BIG, scale, shift, ACT);

  // ---- stage 2: conv2 -> stats -> BN+ReLU+pool -> a2 bf16 [16,32,16^3]
  k_conv_wmma<unsigned short, 16, 32, 32, 14>
      <<<16 * 32 * 32 * 2, 32, 0, stream>>>(ACT, w2bf, c2b, BIG);
  unsigned short* A2ACT = ACT + (size_t)16 * 16 * 32768 / 2;  // reuse tail half
  k_zero<<<1, 128, 0, stream>>>(stats, 128);
  k_stats<<<dim3(32, 32), 256, 0, stream>>>(BIG, sum, sq, 16, 32, 32768);
  k_bnparams<<<1, 64, 0, stream>>>(sum, sq, bn2g, bn2b, scale, shift, 32,
                                   1.0f / (16.0f * 32768.0f));
  k_bnrelupool<32, 32>
      <<<(16 * 32 * 4096 + 255) / 256, 256, 0, stream>>>(BIG, scale, shift, A2ACT);

  // ---- stage 3: conv3 -> stats -> BN (no relu, f32 in place)
  k_conv_wmma<unsigned short, 32, 16, 64, 28>
      <<<16 * 16 * 16, 32, 0, stream>>>(A2ACT, w3bf, c3b, BIG);
  k_zero<<<1, 128, 0, stream>>>(stats, 128);
  k_stats<<<dim3(64, 32), 256, 0, stream>>>(BIG, sum, sq, 16, 64, 4096);
  k_bnparams<<<1, 64, 0, stream>>>(sum, sq, bn3g, bn3b, scale, shift, 64,
                                   1.0f / (16.0f * 4096.0f));
  k_bn_inplace<<<(4194304 + 255) / 256, 256, 0, stream>>>(BIG, scale, shift, 64, 4096,
                                                          4194304);

  // ---- ROI adaptive-avg pool -> bf16 [16,4096]
  k_roipool<<<(65536 + 255) / 256, 256, 0, stream>>>(BIG, prop, A0);

  // ---- FC head (WMMA, M = batch of 16)
  k_fc_wmma<<<1024 / 16, 32, 0, stream>>>(A0, fw0, fcb[0], F, 4096, 1024);
  k_fcbn<<<(1024 + 255) / 256, 256, 0, stream>>>(F, fbg[0], fbb[0], A1, 1024);
  k_fc_wmma<<<512 / 16, 32, 0, stream>>>(A1, fw1, fcb[1], F, 1024, 512);
  k_fcbn<<<(512 + 255) / 256, 256, 0, stream>>>(F, fbg[1], fbb[1], A2, 512);
  k_fc_wmma<<<256 / 16, 32, 0, stream>>>(A2, fw2, fcb[2], F, 512, 256);
  k_fcbn<<<(256 + 255) / 256, 256, 0, stream>>>(F, fbg[2], fbb[2], A3, 256);
  k_fc_wmma<<<128 / 16, 32, 0, stream>>>(A3, fw3, fcb[3], F, 256, 128);
  k_fcbn<<<(128 + 255) / 256, 256, 0, stream>>>(F, fbg[3], fbb[3], A4, 128);
  k_fc_wmma<<<64 / 16, 32, 0, stream>>>(A4, fw4, fcb[4], (float*)d_out, 128, 64);
}
